// EGNN_36352603193957
// MI455X (gfx1250) — matrix operand order repaired and checked by reading, not compiled, not run on hardware
//
#include <hip/hip_runtime.h>
#include <cstdint>
#include <cstddef>

// ---------------- problem constants (from reference) ----------------
#define N_NODES 20000
#define N_EDGES 640000
#define HDIM    128
#define INC     64
#define OUTC    64
#define NLAYERS 4

// ---------------- types ----------------
typedef __bf16 bf16_t;
typedef __attribute__((ext_vector_type(16))) __bf16 v16bf;
typedef __attribute__((ext_vector_type(8)))  __bf16 v8bf;
typedef __attribute__((ext_vector_type(8)))  float  v8f;

union BFU { v16bf v; v8bf h[2]; };

__device__ __forceinline__ float silu_f(float x) { return x / (1.0f + __expf(-x)); }

__device__ __forceinline__ void atomic_add_f32(float* p, float v) {
  __hip_atomic_fetch_add(p, v, __ATOMIC_RELAXED, __HIP_MEMORY_SCOPE_AGENT);
}

template<int NACC>
__device__ __forceinline__ void acc_zero(v8f* acc) {
#pragma unroll
  for (int a = 0; a < NACC; ++a)
#pragma unroll
    for (int i = 0; i < 8; ++i) acc[a][i] = 0.0f;
}

__device__ __forceinline__ BFU load_b_tile(const bf16_t* Bp, int tile, int lane) {
  const bf16_t* bp = Bp + (((size_t)tile * 32 + (size_t)lane) << 4);
  BFU b;
  b.h[0] = *(const v8bf*)(bp);
  b.h[1] = *(const v8bf*)(bp + 8);
  return b;
}

// MT x (16 x K) row-major LDS A-tiles  x  pre-packed bf16 B  ->  MT x 16 x (NT*16) f32 acc.
// A layout: lane L holds row M=L%16; element j -> K=kc*32+(L/16)*16+j (matches packed B)
//  => two contiguous 16B LDS loads per M-tile.
// B pre-packed so lane L reads 32 contiguous bytes: b[j] = B[kc*32+(L/16)*16+j][nt*16+L%16].
// MT register blocking: each B load feeds MT WMMAs.  nt unrolled by 2 with two named B
// temporaries so two B tiles stay in flight (partial s_wait_loadcnt instead of full drains).
template<int KC, int NT, int MT>
__device__ __forceinline__ void wmma_gemm(const bf16_t* Abase, int rowStride,
                                          const bf16_t* Bp, int lane,
                                          v8f* acc /* [MT*NT] */) {
  static_assert((NT & 1) == 0, "NT must be even");
  const int kh8 = (lane >> 4) << 3;    // 0 or 8
  const int r   = lane & 15;
#pragma unroll
  for (int kc = 0; kc < KC; ++kc) {
    BFU a[MT];
#pragma unroll
    for (int mt = 0; mt < MT; ++mt) {
      const bf16_t* Ar = Abase + (size_t)(mt * 16 + r) * rowStride + kc * 32;
      a[mt].h[0] = *(const v8bf*)(Ar + 2 * kh8);
      a[mt].h[1] = *(const v8bf*)(Ar + 2 * kh8 + 8);
    }
#pragma unroll
    for (int nt = 0; nt < NT; nt += 2) {
      BFU b0 = load_b_tile(Bp, kc * NT + nt,     lane);
      BFU b1 = load_b_tile(Bp, kc * NT + nt + 1, lane);
#pragma unroll
      for (int mt = 0; mt < MT; ++mt)
        acc[mt * NT + nt] = __builtin_amdgcn_wmma_f32_16x16x32_bf16(
            false, a[mt].v, false, b0.v, (short)0, acc[mt * NT + nt], false, false);
#pragma unroll
      for (int mt = 0; mt < MT; ++mt)
        acc[mt * NT + nt + 1] = __builtin_amdgcn_wmma_f32_16x16x32_bf16(
            false, a[mt].v, false, b1.v, (short)0, acc[mt * NT + nt + 1], false, false);
    }
  }
}

// C layout: lane L covers N = L%16, M = mt*16 + (L/16)*8 + v.
template<int NT, int MT, bool SILU>
__device__ __forceinline__ void bias_act(v8f* acc, const float* bias, int lane) {
  const int n0 = lane & 15;
#pragma unroll
  for (int nt = 0; nt < NT; ++nt) {
    float bv = bias[nt * 16 + n0];
#pragma unroll
    for (int mt = 0; mt < MT; ++mt)
#pragma unroll
      for (int v = 0; v < 8; ++v) {
        float f = acc[mt * NT + nt][v] + bv;
        acc[mt * NT + nt][v] = SILU ? silu_f(f) : f;
      }
  }
}

template<int NT, int MT>
__device__ __forceinline__ void store_c_bf16_lds(bf16_t* buf, int stride,
                                                 const v8f* acc, int lane) {
  const int n0 = lane & 15, mb = (lane >> 4) << 3;
#pragma unroll
  for (int mt = 0; mt < MT; ++mt)
#pragma unroll
    for (int nt = 0; nt < NT; ++nt)
#pragma unroll
      for (int v = 0; v < 8; ++v)
        buf[(mt * 16 + mb + v) * stride + nt * 16 + n0] = (bf16_t)acc[mt * NT + nt][v];
}

// ---------------- weight repack: fp32 [K x Nc] -> packed bf16 B-operand tiles ----------------
__global__ void egnn_repack(const float* __restrict__ W, bf16_t* __restrict__ P,
                            int K, int Kpad, int Nc) {
  const int idx = blockIdx.x * blockDim.x + threadIdx.x;
  const int total = Kpad * Nc;
  if (idx >= total) return;
  const int NT = Nc >> 4;
  const int j  = idx & 15;
  const int L  = (idx >> 4) & 31;
  const int nt = (idx >> 9) % NT;
  const int kc = idx / (NT << 9);
  const int k  = kc * 32 + ((L >> 4) << 4) + j;
  const int n  = nt * 16 + (L & 15);
  const float v = (k < K) ? W[(size_t)k * Nc + n] : 0.0f;
  P[idx] = (bf16_t)v;
}

// ---------------- fused per-edge MLP chain + scatter (32 edges per wave) ----------------
#define EW 2   // waves (32-edge tiles) per block
__global__ __launch_bounds__(EW * 32) void egnn_edge(
    const bf16_t* __restrict__ hbf, const float* __restrict__ pos,
    const int* __restrict__ eidx, const float* __restrict__ eattr,
    const bf16_t* __restrict__ pW1, const float* __restrict__ b1,
    const bf16_t* __restrict__ pW2, const float* __restrict__ b2,
    const bf16_t* __restrict__ pC1, const float* __restrict__ cb,
    const float* __restrict__ cw2,
    float* __restrict__ agg_feat, float* __restrict__ agg_coord) {
  __shared__ __align__(16) bf16_t ein [EW][32][288];
  __shared__ __align__(16) bf16_t mbuf[EW][32][128];
  __shared__ float cdls[EW][32][3];
  __shared__ int   didx[EW][32];

  const int lane = threadIdx.x & 31;
  const int w    = threadIdx.x >> 5;
  const int e0   = (blockIdx.x * EW + w) * 32;

  // per-edge scalars + h-row gather: one edge per lane, indices stay in registers
  {
    const int e = e0 + lane;
    const int s = eidx[e];              // edge_index[0] = src
    const int d = eidx[N_EDGES + e];    // edge_index[1] = dst
    didx[w][lane] = d;
    __builtin_prefetch(hbf + (size_t)d * HDIM, 0, 3);
    __builtin_prefetch(hbf + (size_t)s * HDIM, 0, 3);
    const float dx = pos[3*s+0] - pos[3*d+0];
    const float dy = pos[3*s+1] - pos[3*d+1];
    const float dz = pos[3*s+2] - pos[3*d+2];
    cdls[w][lane][0] = dx; cdls[w][lane][1] = dy; cdls[w][lane][2] = dz;
    ein[w][lane][256] = (bf16_t)(dx*dx + dy*dy + dz*dz);
    ein[w][lane][257] = (bf16_t)eattr[e];
#pragma unroll
    for (int k = 258; k < 288; ++k) ein[w][lane][k] = (bf16_t)0.0f;
    // cols 0..127 = h[dst], 128..255 = h[src]
    const uint4* sd = (const uint4*)(hbf + (size_t)d * HDIM);
    const uint4* ss = (const uint4*)(hbf + (size_t)s * HDIM);
    uint4* p0 = (uint4*)(&ein[w][lane][0]);
    uint4* p1 = (uint4*)(&ein[w][lane][128]);
#pragma unroll
    for (int i = 0; i < 16; ++i) { p0[i] = sd[i]; p1[i] = ss[i]; }
  }
  __syncthreads();

  v8f acc[16];
  // stage 1: ein[32x288] @ eW1 -> silu -> m1
  acc_zero<16>(acc);
  wmma_gemm<9, 8, 2>(&ein[w][0][0], 288, pW1, lane, acc);
  bias_act<8, 2, true>(acc, b1, lane);
  store_c_bf16_lds<8, 2>(&mbuf[w][0][0], 128, acc, lane);
  __syncthreads();

  // stage 2: m1 @ eW2 -> silu -> m
  acc_zero<16>(acc);
  wmma_gemm<4, 8, 2>(&mbuf[w][0][0], 128, pW2, lane, acc);
  bias_act<8, 2, true>(acc, b2, lane);
  __syncthreads();
  store_c_bf16_lds<8, 2>(&mbuf[w][0][0], 128, acc, lane);   // m (bf16) for stage 3
  __syncthreads();

  // segment_sum(m, dst): f32 atomics into agg_feat (L2-resident)
  {
    const int n0 = lane & 15, mb = (lane >> 4) << 3;
#pragma unroll
    for (int mt = 0; mt < 2; ++mt)
#pragma unroll
      for (int v = 0; v < 8; ++v) {
        const int d = didx[w][mt * 16 + mb + v];
        float* base = agg_feat + (size_t)d * HDIM + n0;
#pragma unroll
        for (int nt = 0; nt < 8; ++nt)
          atomic_add_f32(base + nt * 16, acc[mt * 8 + nt][v]);
      }
  }

  // stage 3: m @ cW1 -> silu (stored into ein cols 0..127, stride 288)
  acc_zero<16>(acc);
  wmma_gemm<4, 8, 2>(&mbuf[w][0][0], 128, pC1, lane, acc);
  bias_act<8, 2, true>(acc, cb, lane);
  store_c_bf16_lds<8, 2>(&ein[w][0][0], 288, acc, lane);
  __syncthreads();

  // dot with cW2 and coordinate scatter (all 32 lanes, one edge each)
  {
    float s = 0.0f;
    for (int k = 0; k < 128; ++k) s += (float)ein[w][lane][k] * cw2[k];
    const int d = didx[w][lane];
    atomic_add_f32(&agg_coord[3*d+0], cdls[w][lane][0] * s);
    atomic_add_f32(&agg_coord[3*d+1], cdls[w][lane][1] * s);
    atomic_add_f32(&agg_coord[3*d+2], cdls[w][lane][2] * s);
  }
}

// ---------------- node update: h = silu([h|agg]@nW1+b)@nW2+b ; pos += agg_coord ----------------
#define NW 2   // waves (16-node tiles) per block
__global__ __launch_bounds__(NW * 32) void egnn_node(
    const bf16_t* __restrict__ hbf, const float* __restrict__ agg_feat,
    const float* __restrict__ agg_coord,
    const bf16_t* __restrict__ pW1, const float* __restrict__ b1,
    const bf16_t* __restrict__ pW2, const float* __restrict__ b2,
    float* __restrict__ pos, bf16_t* __restrict__ hout) {
  __shared__ __align__(16) bf16_t A  [NW][16][256];
  __shared__ __align__(16) bf16_t mb2[NW][16][128];
  const int lane = threadIdx.x & 31;
  const int w    = threadIdx.x >> 5;
  const int r    = lane & 15;
  const int half = lane >> 4;
  const int node = (blockIdx.x * NW + w) * 16 + r;

  if (half == 0) {
    const uint4* s4 = (const uint4*)(hbf + (size_t)node * HDIM);
    uint4* d4 = (uint4*)(&A[w][r][0]);
#pragma unroll
    for (int i = 0; i < 16; ++i) d4[i] = s4[i];
#pragma unroll
    for (int c = 0; c < 3; ++c) pos[3*node+c] += agg_coord[3*node+c];
  } else {
    const float4* f4 = (const float4*)(agg_feat + (size_t)node * HDIM);
#pragma unroll
    for (int i = 0; i < 32; ++i) {
      float4 f = f4[i];
      A[w][r][128 + 4*i + 0] = (bf16_t)f.x;
      A[w][r][128 + 4*i + 1] = (bf16_t)f.y;
      A[w][r][128 + 4*i + 2] = (bf16_t)f.z;
      A[w][r][128 + 4*i + 3] = (bf16_t)f.w;
    }
  }
  __syncthreads();

  v8f acc[8];
  acc_zero<8>(acc);
  wmma_gemm<8, 8, 1>(&A[w][0][0], 256, pW1, lane, acc);
  bias_act<8, 1, true>(acc, b1, lane);
  store_c_bf16_lds<8, 1>(&mb2[w][0][0], 128, acc, lane);
  __syncthreads();

  acc_zero<8>(acc);
  wmma_gemm<4, 8, 1>(&mb2[w][0][0], 128, pW2, lane, acc);
  bias_act<8, 1, false>(acc, b2, lane);
  store_c_bf16_lds<8, 1>(&A[w][0][0], 256, acc, lane);   // result into cols 0..127
  __syncthreads();
  {
    const uint4* s4 = (const uint4*)(&A[w][r][half * 64]);
    uint4* d4 = (uint4*)(hout + (size_t)node * HDIM + half * 64);
#pragma unroll
    for (int i = 0; i < 8; ++i) d4[i] = s4[i];
  }
}

// ---------------- input embedding: h0 = x @ Win + bin ----------------
__global__ __launch_bounds__(NW * 32) void egnn_embed(
    const float* __restrict__ x, const bf16_t* __restrict__ pWin,
    const float* __restrict__ bin, bf16_t* __restrict__ hout) {
  __shared__ __align__(16) bf16_t A [NW][16][64];
  __shared__ __align__(16) bf16_t Ob[NW][16][128];
  const int lane = threadIdx.x & 31;
  const int w    = threadIdx.x >> 5;
  const int r    = lane & 15;
  const int half = lane >> 4;
  const int node = (blockIdx.x * NW + w) * 16 + r;
  {
    const float4* f4 = (const float4*)(x + (size_t)node * INC + half * 32);
#pragma unroll
    for (int i = 0; i < 8; ++i) {
      float4 f = f4[i];
      A[w][r][half*32 + 4*i + 0] = (bf16_t)f.x;
      A[w][r][half*32 + 4*i + 1] = (bf16_t)f.y;
      A[w][r][half*32 + 4*i + 2] = (bf16_t)f.z;
      A[w][r][half*32 + 4*i + 3] = (bf16_t)f.w;
    }
  }
  __syncthreads();
  v8f acc[8];
  acc_zero<8>(acc);
  wmma_gemm<2, 8, 1>(&A[w][0][0], 64, pWin, lane, acc);
  bias_act<8, 1, false>(acc, bin, lane);
  store_c_bf16_lds<8, 1>(&Ob[w][0][0], 128, acc, lane);
  __syncthreads();
  {
    const uint4* s4 = (const uint4*)(&Ob[w][r][half * 64]);
    uint4* d4 = (uint4*)(hout + (size_t)node * HDIM + half * 64);
#pragma unroll
    for (int i = 0; i < 8; ++i) d4[i] = s4[i];
  }
}

// ---------------- output: out = h @ Wout + bout ; append pos ----------------
__global__ __launch_bounds__(NW * 32) void egnn_out(
    const bf16_t* __restrict__ hbf, const bf16_t* __restrict__ pWout,
    const float* __restrict__ bout, const float* __restrict__ pos,
    float* __restrict__ out) {
  __shared__ __align__(16) bf16_t A [NW][16][128];
  __shared__ __align__(16) float  Co[NW][16][64];
  const int lane = threadIdx.x & 31;
  const int w    = threadIdx.x >> 5;
  const int r    = lane & 15;
  const int half = lane >> 4;
  const int node = (blockIdx.x * NW + w) * 16 + r;
  {
    const uint4* s4 = (const uint4*)(hbf + (size_t)node * HDIM + half * 64);
    uint4* d4 = (uint4*)(&A[w][r][half * 64]);
#pragma unroll
    for (int i = 0; i < 8; ++i) d4[i] = s4[i];
  }
  __syncthreads();
  v8f acc[4];
  acc_zero<4>(acc);
  wmma_gemm<4, 4, 1>(&A[w][0][0], 128, pWout, lane, acc);
  bias_act<4, 1, false>(acc, bout, lane);
  {
    const int n0 = lane & 15, mb = (lane >> 4) << 3;
#pragma unroll
    for (int nt = 0; nt < 4; ++nt)
#pragma unroll
      for (int v = 0; v < 8; ++v)
        Co[w][mb + v][nt * 16 + n0] = acc[nt][v];
  }
  __syncthreads();
  {
    const float4* s4 = (const float4*)(&Co[w][r][half * 32]);
    float4* d4 = (float4*)(out + (size_t)node * OUTC + half * 32);
#pragma unroll
    for (int i = 0; i < 8; ++i) d4[i] = s4[i];
  }
  if (half == 0) {
    float* opos = out + (size_t)N_NODES * OUTC;
#pragma unroll
    for (int c = 0; c < 3; ++c) opos[3*node+c] = pos[3*node+c];
  }
}

// ---------------- host ----------------
extern "C" void kernel_launch(void* const* d_in, const int* in_sizes, int n_in,
                              void* d_out, int out_size, void* d_ws, size_t ws_size,
                              hipStream_t stream) {
  const float* x     = (const float*)d_in[0];
  const float* pos0  = (const float*)d_in[1];
  const int*   eidx  = (const int*)d_in[2];
  const float* eattr = (const float*)d_in[3];
  const float* Win   = (const float*)d_in[4];
  const float* bin   = (const float*)d_in[5];
  const float* Wout  = (const float*)d_in[6];
  const float* bout  = (const float*)d_in[7];
  const float* eW1   = (const float*)d_in[8];
  const float* eb1   = (const float*)d_in[9];
  const float* eW2   = (const float*)d_in[10];
  const float* eb2   = (const float*)d_in[11];
  const float* nW1   = (const float*)d_in[12];
  const float* nb1   = (const float*)d_in[13];
  const float* nW2   = (const float*)d_in[14];
  const float* nb2   = (const float*)d_in[15];
  const float* cW1   = (const float*)d_in[16];
  const float* cb1   = (const float*)d_in[17];
  const float* cW2   = (const float*)d_in[18];

  uint8_t* wp = (uint8_t*)d_ws;
  auto carve = [&](size_t bytes) -> void* {
    void* p = (void*)wp;
    wp += (bytes + 255) & ~(size_t)255;
    return p;
  };
  float*  pos_ws    = (float*)carve((size_t)N_NODES * 3 * 4);
  float*  agg_feat  = (float*)carve((size_t)N_NODES * HDIM * 4);
  float*  agg_coord = (float*)carve((size_t)N_NODES * 3 * 4);
  bf16_t* hA    = (bf16_t*)carve((size_t)N_NODES * HDIM * 2);
  bf16_t* hB    = (bf16_t*)carve((size_t)N_NODES * HDIM * 2);
  bf16_t* pWin  = (bf16_t*)carve((size_t)INC * HDIM * 2);
  bf16_t* peW1  = (bf16_t*)carve((size_t)NLAYERS * 288 * HDIM * 2);
  bf16_t* peW2  = (bf16_t*)carve((size_t)NLAYERS * HDIM * HDIM * 2);
  bf16_t* pnW1  = (bf16_t*)carve((size_t)NLAYERS * 256 * HDIM * 2);
  bf16_t* pnW2  = (bf16_t*)carve((size_t)NLAYERS * HDIM * HDIM * 2);
  bf16_t* pcW1  = (bf16_t*)carve((size_t)NLAYERS * HDIM * HDIM * 2);
  bf16_t* pWout = (bf16_t*)carve((size_t)HDIM * OUTC * 2);

  auto repack = [&](const float* src, bf16_t* dst, int K, int Kpad, int Nc) {
    const int total = Kpad * Nc;
    egnn_repack<<<(total + 255) / 256, 256, 0, stream>>>(src, dst, K, Kpad, Nc);
  };
  repack(Win,  pWin,  64, 64, 128);
  repack(Wout, pWout, 128, 128, 64);
  for (int l = 0; l < NLAYERS; ++l) {
    repack(eW1 + (size_t)l*258*128, peW1 + (size_t)l*288*128, 258, 288, 128);
    repack(eW2 + (size_t)l*128*128, peW2 + (size_t)l*128*128, 128, 128, 128);
    repack(nW1 + (size_t)l*256*128, pnW1 + (size_t)l*256*128, 256, 256, 128);
    repack(nW2 + (size_t)l*128*128, pnW2 + (size_t)l*128*128, 128, 128, 128);
    repack(cW1 + (size_t)l*128*128, pcW1 + (size_t)l*128*128, 128, 128, 128);
  }

  hipMemcpyAsync(pos_ws, pos0, (size_t)N_NODES * 3 * 4, hipMemcpyDeviceToDevice, stream);

  egnn_embed<<<N_NODES / (16 * NW), NW * 32, 0, stream>>>(x, pWin, bin, hA);

  bf16_t* hcur = hA;
  bf16_t* hnxt = hB;
  for (int l = 0; l < NLAYERS; ++l) {
    hipMemsetAsync(agg_feat, 0, (size_t)N_NODES * HDIM * 4, stream);
    hipMemsetAsync(agg_coord, 0, (size_t)N_NODES * 3 * 4, stream);
    egnn_edge<<<N_EDGES / (32 * EW), EW * 32, 0, stream>>>(
        hcur, pos_ws, eidx, eattr,
        peW1 + (size_t)l*288*128, eb1 + l*128,
        peW2 + (size_t)l*128*128, eb2 + l*128,
        pcW1 + (size_t)l*128*128, cb1 + l*128,
        cW2 + l*128,
        agg_feat, agg_coord);
    egnn_node<<<N_NODES / (16 * NW), NW * 32, 0, stream>>>(
        hcur, agg_feat, agg_coord,
        pnW1 + (size_t)l*256*128, nb1 + l*128,
        pnW2 + (size_t)l*128*128, nb2 + l*128,
        pos_ws, hnxt);
    bf16_t* t = hcur; hcur = hnxt; hnxt = t;
  }

  egnn_out<<<N_NODES / (16 * NW), NW * 32, 0, stream>>>(
      hcur, pWout, bout, pos_ws, (float*)d_out);

  (void)in_sizes; (void)n_in; (void)out_size; (void)ws_size;
}
